// SpaceTempGoG_detr_dota_16277926052234
// MI455X (gfx1250) — matrix operand order repaired and checked by reading, not compiled, not run on hardware
//
#include <hip/hip_runtime.h>
#include <cstddef>

// ---------------------------------------------------------------------------
// Model constants (match reference)
// ---------------------------------------------------------------------------
#define NNODE 10000
#define TGRAPH 100
#define NPG 100
#define NEDGE 100000
#define HEADS 4
#define CHEAD 128
#define FGAT 512           // HEADS*CHEAD
#define DIN 2048
#define DLBL 300
#define K1 80
#define K2 56
#define HD 512

// ---------------------------------------------------------------------------
// Types for WMMA and TDM descriptors
// ---------------------------------------------------------------------------
typedef __attribute__((ext_vector_type(16))) _Float16 v16h;
typedef __attribute__((ext_vector_type(8)))  float    v8f;
typedef unsigned int v4u __attribute__((ext_vector_type(4)));
typedef int          v8i __attribute__((ext_vector_type(8)));
typedef int          v4i __attribute__((ext_vector_type(4)));

#define TBM 128
#define TBN 64
#define TBK 32

// ---------------------------------------------------------------------------
// TDM: issue a 2D tile load (global f32 -> LDS) via Tensor Data Mover.
// D# packing per CDNA5 ISA ch.8: group0 = {flags/count, lds_addr, gaddr, type},
// group1 = {data_size, tensor dims, tile dims, dim0 stride}. Tensor dims are
// the *remaining* extents from the tile origin so edge tiles zero-fill.
// This toolchain exposes the 6-arg builtin (g0, g1, g2, g3, g4, cpol).
// ---------------------------------------------------------------------------
__device__ inline void tdm_load_2d(const void* gptr, unsigned lds_off,
                                   unsigned tdim0, unsigned tdim1,
                                   unsigned tile0, unsigned tile1,
                                   unsigned long long stride0)
{
  unsigned long long ga = (unsigned long long)(size_t)gptr;
  v4u g0;
  g0[0] = 1u;                                            // count=1, load, user D#
  g0[1] = lds_off;                                       // LDS byte address
  g0[2] = (unsigned)(ga & 0xFFFFFFFFu);                  // global_addr[31:0]
  g0[3] = (unsigned)((ga >> 32) & 0x01FFFFFFu)           // global_addr[56:32]
        | (2u << 30);                                    // type = 2 ("image")
  v8i g1;
  g1[0] = (int)(2u << 16);                               // data_size = 4 bytes
  g1[1] = (int)((tdim0 & 0xFFFFu) << 16);                // tensor_dim0[15:0]
  g1[2] = (int)(((tdim0 >> 16) & 0xFFFFu)
        | ((tdim1 & 0xFFFFu) << 16));                    // dim0[31:16] | dim1[15:0]
  g1[3] = (int)(((tdim1 >> 16) & 0xFFFFu)
        | ((tile0 & 0xFFFFu) << 16));                    // dim1[31:16] | tile_dim0
  g1[4] = (int)(tile1 & 0xFFFFu);                        // tile_dim1 (tile_dim2=0)
  g1[5] = (int)(unsigned)(stride0 & 0xFFFFFFFFull);      // tensor_dim0_stride lo
  g1[6] = (int)(unsigned)((stride0 >> 32) & 0xFFFFull);  // stride hi (dim1_stride=0)
  g1[7] = 0;
  v4i gz;  gz[0] = 0; gz[1] = 0; gz[2] = 0; gz[3] = 0;   // 2D: groups 2/3 unused
  v8i gz8; gz8[0] = 0; gz8[1] = 0; gz8[2] = 0; gz8[3] = 0;
           gz8[4] = 0; gz8[5] = 0; gz8[6] = 0; gz8[7] = 0;
  __builtin_amdgcn_tensor_load_to_lds(g0, g1, gz, gz, gz8, 0);
}

// ---------------------------------------------------------------------------
// WMMA GEMM with TDM double-buffered staging:
//   C[M,N] = A[M,K](lda) * B[K,N] (+bias[N]) , f32 in/out, f16 WMMA MACs.
// Wave 0 drives the Tensor Data Mover (async tile DMA into LDS f32 buffers,
// tracked by TENSORcnt); all 8 waves convert f32->f16 tiles and run WMMAs.
// ---------------------------------------------------------------------------
__global__ __launch_bounds__(256) void gemm_f16_wmma(
    const float* __restrict__ A, int lda,
    const float* __restrict__ B,
    const float* __restrict__ bias,
    float* __restrict__ C, int ldc,
    int M, int N, int K)
{
  __shared__ float    ldsA32[2][TBM][TBK];   // 2 x 16 KB (TDM destination)
  __shared__ float    ldsB32[2][TBK][TBN];   // 2 x  8 KB (TDM destination)
  __shared__ _Float16 sA[TBM][TBK + 8];      // f16 A tile (padded rows)
  __shared__ _Float16 sB[TBN][TBK + 8];      // f16 B tile, transposed [n][k]

  const int tid  = threadIdx.x;
  const int lane = tid & 31;
  const int wave = tid >> 5;
  const int bm = blockIdx.x * TBM;
  const int bn = blockIdx.y * TBN;
  const int wm = (wave & 3) * 32;
  const int wn = (wave >> 2) * 32;
  const int lrow = lane & 15;
  const int hsel = lane >> 4;

  v8f acc[2][2];
  for (int i = 0; i < 2; ++i)
    for (int j = 0; j < 2; ++j)
      for (int e = 0; e < 8; ++e) acc[i][j][e] = 0.0f;

  const int ntiles = (K + TBK - 1) / TBK;
  const unsigned aoff[2] = {(unsigned)(size_t)&ldsA32[0][0][0],
                            (unsigned)(size_t)&ldsA32[1][0][0]};
  const unsigned boff[2] = {(unsigned)(size_t)&ldsB32[0][0][0],
                            (unsigned)(size_t)&ldsB32[1][0][0]};

  if (wave == 0) {
    // prologue: DMA K-tile 0 into buffer 0
    tdm_load_2d(A + (size_t)bm * lda, aoff[0],
                (unsigned)K, (unsigned)(M - bm), TBK, TBM,
                (unsigned long long)lda);
    tdm_load_2d(B + bn, boff[0],
                (unsigned)(N - bn), (unsigned)K, TBN, TBK,
                (unsigned long long)N);
  }

  for (int t = 0; t < ntiles; ++t) {
    const int buf = t & 1;
    if (wave == 0) {
      if (t + 1 < ntiles) {
        const int k0n = (t + 1) * TBK;
        // post next tile's DMA, then wait for the current tile's pair
        tdm_load_2d(A + (size_t)bm * lda + k0n, aoff[buf ^ 1],
                    (unsigned)(K - k0n), (unsigned)(M - bm), TBK, TBM,
                    (unsigned long long)lda);
        tdm_load_2d(B + (size_t)k0n * N + bn, boff[buf ^ 1],
                    (unsigned)(N - bn), (unsigned)(K - k0n), TBN, TBK,
                    (unsigned long long)N);
        __builtin_amdgcn_s_wait_tensorcnt(2);
      } else {
        __builtin_amdgcn_s_wait_tensorcnt(0);
      }
    }
    __syncthreads();   // current-tile DMA complete & visible to all waves

    // ---- convert landed f32 tiles to f16 (B transposed) ----
    for (int it = 0; it < 4; ++it) {
      int idx = tid + it * 256;            // 1024 float4 slots (128x32)
      int r  = idx >> 3;
      int c4 = (idx & 7) << 2;
      const float4 t4 = *(const float4*)&ldsA32[buf][r][c4];
      sA[r][c4 + 0] = (_Float16)t4.x;
      sA[r][c4 + 1] = (_Float16)t4.y;
      sA[r][c4 + 2] = (_Float16)t4.z;
      sA[r][c4 + 3] = (_Float16)t4.w;
    }
    for (int it = 0; it < 2; ++it) {
      int idx = tid + it * 256;            // 512 float4 slots (32x64)
      int r  = idx >> 4;                   // k row
      int c4 = (idx & 15) << 2;            // n col
      const float4 t4 = *(const float4*)&ldsB32[buf][r][c4];
      sB[c4 + 0][r] = (_Float16)t4.x;
      sB[c4 + 1][r] = (_Float16)t4.y;
      sB[c4 + 2][r] = (_Float16)t4.z;
      sB[c4 + 3][r] = (_Float16)t4.w;
    }
    __syncthreads();

    // ---- fragments per ISA 7.12.2 layouts ----
    union Frag { v16h v; uint4 q[2]; };
    Frag fa[2], fb[2];
    for (int i = 0; i < 2; ++i) {
      const uint4* p = (const uint4*)(&sA[wm + i * 16 + lrow][0]);
      fa[i].q[0] = p[hsel];          // lanes0-15: K0..7   lanes16-31: K8..15
      fa[i].q[1] = p[hsel + 2];      // lanes0-15: K16..23 lanes16-31: K24..31
    }
    for (int j = 0; j < 2; ++j) {
      const uint4* p = (const uint4*)(&sB[wn + j * 16 + lrow][0]);
      fb[j].q[0] = p[hsel];          // lanes0-15: K0..7   lanes16-31: K8..15
      fb[j].q[1] = p[hsel + 2];      // lanes0-15: K16..23 lanes16-31: K24..31
    }
    for (int i = 0; i < 2; ++i)
      for (int j = 0; j < 2; ++j)
        acc[i][j] = __builtin_amdgcn_wmma_f32_16x16x32_f16(
            false, fa[i].v, false, fb[j].v, (short)0, acc[i][j], false, false);
    __syncthreads();
  }

  // ---- epilogue: C layout VGPR e -> (M = e + 8*hsel, N = lane&15) ----
  for (int i = 0; i < 2; ++i)
    for (int j = 0; j < 2; ++j) {
      int colg = bn + wn + j * 16 + lrow;
      if (colg >= N) continue;
      float badd = bias ? bias[colg] : 0.0f;
      for (int e = 0; e < 8; ++e) {
        int rowg = bm + wm + i * 16 + e + hsel * 8;
        if (rowg < M) C[(size_t)rowg * ldc + colg] = acc[i][j][e] + badd;
      }
    }
}

// ---------------------------------------------------------------------------
// Column statistics (sum / sumsq over rows), for BatchNorm / InstanceNorm
// ---------------------------------------------------------------------------
__global__ void colstats(const float* __restrict__ A, int M, int N,
                         float* __restrict__ sums)   // sums[0..N)=sum, [N..2N)=sumsq
{
  int c = blockIdx.x * blockDim.x + threadIdx.x;
  if (c >= N) return;
  int rows_per = (M + gridDim.y - 1) / gridDim.y;
  int r0 = blockIdx.y * rows_per;
  int r1 = r0 + rows_per; if (r1 > M) r1 = M;
  float s = 0.f, s2 = 0.f;
  for (int r = r0; r < r1; ++r) {
    float v = A[(size_t)r * N + c];
    s += v; s2 += v * v;
  }
  if (r1 > r0) { atomicAdd(&sums[c], s); atomicAdd(&sums[N + c], s2); }
}

// normalize (+optional affine) (+optional lrelu); writes into strided slice
__global__ void norm_apply(const float* __restrict__ A, const float* __restrict__ sums,
                           const float* __restrict__ g, const float* __restrict__ b,
                           float* __restrict__ out, int M, int N, int ostride, int ocol0,
                           int flags)  // bit0 lrelu, bit1 affine
{
  size_t gid = (size_t)blockIdx.x * blockDim.x + threadIdx.x;
  if (gid >= (size_t)M * N) return;
  int r = (int)(gid / N), c = (int)(gid % N);
  float mean = sums[c] / (float)M;
  float var  = sums[N + c] / (float)M - mean * mean;
  float v = (A[gid] - mean) * rsqrtf(var + 1e-5f);
  if (flags & 2) v = v * g[c] + b[c];
  if (flags & 1) v = v > 0.f ? v : 0.2f * v;
  out[(size_t)r * ostride + ocol0 + c] = v;
}

// ---------------------------------------------------------------------------
// Small utility kernels
// ---------------------------------------------------------------------------
__global__ void fillf(float* p, float v, size_t n) {
  size_t i = (size_t)blockIdx.x * blockDim.x + threadIdx.x;
  if (i < n) p[i] = v;
}
__global__ void lrelu_inplace(float* p, size_t n) {
  size_t i = (size_t)blockIdx.x * blockDim.x + threadIdx.x;
  if (i < n) { float v = p[i]; p[i] = v > 0.f ? v : 0.2f * v; }
}
__global__ void sigmoid_inplace(float* p, size_t n) {
  size_t i = (size_t)blockIdx.x * blockDim.x + threadIdx.x;
  if (i < n) p[i] = 1.f / (1.f + __expf(-p[i]));
}
__global__ void copy_strided(const float* __restrict__ src, float* __restrict__ dst,
                             int rows, int cols, int sstride, int dstride) {
  size_t gid = (size_t)blockIdx.x * blockDim.x + threadIdx.x;
  if (gid >= (size_t)rows * cols) return;
  int r = (int)(gid / cols), c = (int)(gid % cols);
  dst[(size_t)r * dstride + c] = src[(size_t)r * sstride + c];
}
__global__ void transpose_k(const float* __restrict__ A, float* __restrict__ B,
                            int R, int Cc) {   // A[R,Cc] -> B[Cc,R]
  size_t gid = (size_t)blockIdx.x * blockDim.x + threadIdx.x;
  if (gid >= (size_t)R * Cc) return;
  int r = (int)(gid / Cc), c = (int)(gid % Cc);
  B[(size_t)c * R + r] = A[gid];
}

// ---------------------------------------------------------------------------
// Edge embedding: eemb[e][c] = lrelu(ee[e]*W[c] + b[c])
// ---------------------------------------------------------------------------
__global__ void edge_embed(const float* __restrict__ ee, const float* __restrict__ W,
                           const float* __restrict__ b, float* __restrict__ out, int E_) {
  size_t gid = (size_t)blockIdx.x * blockDim.x + threadIdx.x;
  if (gid >= (size_t)E_ * 128) return;
  int e = (int)(gid >> 7), c = (int)(gid & 127);
  float v = ee[e] * W[c] + b[c];
  out[gid] = v > 0.f ? v : 0.2f * v;
}
// ea_mean[j] = mean_e(eemb[e]) @ We   (from column sums of eemb)
__global__ void ea_mean_k(const float* __restrict__ esum, const float* __restrict__ We,
                          float inv_e, float* __restrict__ out) {
  int j = blockIdx.x * blockDim.x + threadIdx.x;
  if (j >= FGAT) return;
  float acc = 0.f;
  for (int c = 0; c < 128; ++c) acc += (esum[c] * inv_e) * We[(size_t)c * FGAT + j];
  out[j] = acc;
}

// ---------------------------------------------------------------------------
// GATv2 edge kernels
// ---------------------------------------------------------------------------
__device__ inline void atomicMaxFloat(float* addr, float val) {
  if (val >= 0.f) atomicMax((int*)addr, __float_as_int(val));
  else            atomicMin((unsigned int*)addr, __float_as_uint(val));
}

// wave-per-edge: logit[e][h] = sum_c att[h,c]*lrelu(xl[src]+xr[dst]+ea[e]); segment-max
__global__ __launch_bounds__(256) void gat_logits(
    const int* __restrict__ srcI, const int* __restrict__ dstI, int E_, int Nn,
    const float* __restrict__ xl, const float* __restrict__ xr,
    const float* __restrict__ ea, const float* __restrict__ ea_mean,
    const float* __restrict__ att, float* __restrict__ logit, float* __restrict__ mnode)
{
  int wave = (blockIdx.x * blockDim.x + threadIdx.x) >> 5;
  int lane = threadIdx.x & 31;
  int EN = E_ + Nn;
  if (wave >= EN) return;
  int s = wave < E_ ? srcI[wave] : (wave - E_);
  int d = wave < E_ ? dstI[wave] : (wave - E_);
  const float* eap = wave < E_ ? (ea + (size_t)wave * FGAT) : ea_mean;
  const float* xlp = xl + (size_t)s * FGAT;
  const float* xrp = xr + (size_t)d * FGAT;
  for (int h = 0; h < HEADS; ++h) {
    float p = 0.f;
    for (int cc = 0; cc < 4; ++cc) {
      int c = h * CHEAD + lane + cc * 32;
      float v = xlp[c] + xrp[c] + eap[c];
      v = v > 0.f ? v : 0.2f * v;
      p += v * att[c];
    }
    for (int off = 16; off; off >>= 1) p += __shfl_xor(p, off, 32);
    if (lane == 0) {
      logit[(size_t)wave * HEADS + h] = p;
      atomicMaxFloat(&mnode[(size_t)d * HEADS + h], p);
    }
  }
}

__global__ void gat_expsum(const int* __restrict__ dstI, int E_, int Nn,
                           const float* __restrict__ logit, const float* __restrict__ mnode,
                           float* __restrict__ a, float* __restrict__ snode)
{
  size_t gid = (size_t)blockIdx.x * blockDim.x + threadIdx.x;
  int EN = E_ + Nn;
  if (gid >= (size_t)EN * HEADS) return;
  int e = (int)(gid >> 2), h = (int)(gid & 3);
  int d = e < E_ ? dstI[e] : (e - E_);
  float av = __expf(logit[gid] - mnode[(size_t)d * HEADS + h]);
  a[gid] = av;
  atomicAdd(&snode[(size_t)d * HEADS + h], av);
}

__global__ __launch_bounds__(256) void gat_aggregate(
    const int* __restrict__ srcI, const int* __restrict__ dstI, int E_, int Nn,
    const float* __restrict__ xl, const float* __restrict__ a,
    const float* __restrict__ snode, float* __restrict__ out)
{
  int e = blockIdx.x;
  int s = e < E_ ? srcI[e] : (e - E_);
  int d = e < E_ ? dstI[e] : (e - E_);
  for (int c = threadIdx.x; c < FGAT; c += 256) {
    int h = c >> 7;
    float alpha = a[(size_t)e * HEADS + h] / (snode[(size_t)d * HEADS + h] + 1e-16f);
    atomicAdd(&out[(size_t)d * FGAT + c], alpha * xl[(size_t)s * FGAT + c]);
  }
}

// ---------------------------------------------------------------------------
// TopK pooling
// ---------------------------------------------------------------------------
__global__ void inv_norm(const float* __restrict__ w, int n, float* out) {
  __shared__ float sh[256];
  float s = 0.f;
  for (int i = threadIdx.x; i < n; i += 256) s += w[i] * w[i];
  sh[threadIdx.x] = s; __syncthreads();
  for (int st = 128; st; st >>= 1) {
    if ((int)threadIdx.x < st) sh[threadIdx.x] += sh[threadIdx.x + st];
    __syncthreads();
  }
  if (threadIdx.x == 0) out[0] = rsqrtf(sh[0]);
}

__global__ void topk_scores(const float* __restrict__ xg, const float* __restrict__ w,
                            const float* __restrict__ invn, int n, float* __restrict__ sc) {
  int g = blockIdx.x, node = threadIdx.x;
  if (node >= n) return;
  const float* row = xg + ((size_t)g * n + node) * 1024;
  float s = 0.f;
  for (int c = 0; c < 1024; ++c) s += row[c] * w[c];
  sc[(size_t)g * n + node] = tanhf(s * invn[0]);
}

__global__ __launch_bounds__(256) void topk_select(
    const float* __restrict__ xg, const float* __restrict__ sc,
    int n, int k, float* __restrict__ out)
{
  int g = blockIdx.x;
  __shared__ float s[128];
  __shared__ int taken[128];
  __shared__ int selIdx; __shared__ float selVal;
  for (int i = threadIdx.x; i < n; i += blockDim.x) { s[i] = sc[(size_t)g * n + i]; taken[i] = 0; }
  __syncthreads();
  for (int iter = 0; iter < k; ++iter) {
    if (threadIdx.x == 0) {
      float best = -1e30f; int bi = 0;
      for (int i = 0; i < n; ++i)
        if (!taken[i] && s[i] > best) { best = s[i]; bi = i; }
      taken[bi] = 1; selIdx = bi; selVal = best;
    }
    __syncthreads();
    const float* src = xg + ((size_t)g * n + selIdx) * 1024;
    float* dst = out + ((size_t)g * k + iter) * 1024;
    float v = selVal;
    for (int c = threadIdx.x; c < 1024; c += blockDim.x) dst[c] = src[c] * v;
    __syncthreads();
  }
}

__global__ __launch_bounds__(256) void graph_readout(const float* __restrict__ xg, int n,
                                                     float* __restrict__ g) {
  int t = blockIdx.x;
  for (int c = threadIdx.x; c < 1024; c += blockDim.x) {
    float mx = -1e30f, sm = 0.f;
    for (int i = 0; i < n; ++i) {
      float v = xg[((size_t)t * n + i) * 1024 + c];
      mx = fmaxf(mx, v); sm += v;
    }
    g[(size_t)t * 2048 + c] = mx;
    g[(size_t)t * 2048 + 1024 + c] = sm / (float)n;
  }
}

// ---------------------------------------------------------------------------
// Sequential GRU layer (input gates gi precomputed by GEMM: gi=[T,768])
// ---------------------------------------------------------------------------
__global__ __launch_bounds__(256) void gru_seq(
    const float* __restrict__ gi, const float* __restrict__ WhhT /*[256,768]*/,
    const float* __restrict__ bhh, float* __restrict__ ys, int T_)
{
  __shared__ float h[256];
  __shared__ float gh[768];
  int t = threadIdx.x;
  h[t] = 0.f;
  __syncthreads();
  for (int step = 0; step < T_; ++step) {
    for (int o = t; o < 768; o += 256) {
      float acc = bhh[o];
      for (int kk = 0; kk < 256; ++kk) acc += WhhT[(size_t)kk * 768 + o] * h[kk];
      gh[o] = acc;
    }
    __syncthreads();
    const float* gip = gi + (size_t)step * 768;
    float r = 1.f / (1.f + __expf(-(gip[t]       + gh[t])));
    float z = 1.f / (1.f + __expf(-(gip[256 + t] + gh[256 + t])));
    float cand = tanhf(gip[512 + t] + r * gh[512 + t]);
    float h2 = (1.f - z) * cand + z * h[t];
    __syncthreads();
    h[t] = h2;
    ys[(size_t)step * 256 + t] = h2;
    __syncthreads();
  }
}

// ---------------------------------------------------------------------------
// Cross attention: block per (query i, head h)
// ---------------------------------------------------------------------------
__global__ __launch_bounds__(128) void cross_attn(
    const float* __restrict__ q, const float* __restrict__ k,
    const float* __restrict__ v, float* __restrict__ ao, int T_)
{
  int i = blockIdx.x, h = blockIdx.y, t = threadIdx.x;
  __shared__ float qv[HD];
  __shared__ float sc[128];
  __shared__ float ssum;
  for (int c = t; c < HD; c += 128) qv[c] = q[(size_t)i * 2048 + h * HD + c];
  __syncthreads();
  if (t < T_) {
    float s = 0.f;
    const float* kp = k + (size_t)t * 2048 + h * HD;
    for (int c = 0; c < HD; ++c) s += qv[c] * kp[c];
    sc[t] = s * 0.04419417382415922f;   // 1/sqrt(512)
  }
  __syncthreads();
  if (t == 0) {
    float m = -1e30f;
    for (int j = 0; j < T_; ++j) m = fmaxf(m, sc[j]);
    float su = 0.f;
    for (int j = 0; j < T_; ++j) { sc[j] = __expf(sc[j] - m); su += sc[j]; }
    ssum = su;
  }
  __syncthreads();
  float inv = 1.f / ssum;
  for (int c = t; c < HD; c += 128) {
    float acc = 0.f;
    for (int j = 0; j < T_; ++j) acc += sc[j] * v[(size_t)j * 2048 + h * HD + c];
    ao[(size_t)i * 2048 + h * HD + c] = acc * inv;
  }
}

__global__ void softmax2(const float* __restrict__ lg, float* __restrict__ out, int T_) {
  int r = blockIdx.x * blockDim.x + threadIdx.x;
  if (r >= T_) return;
  float a = lg[r * 2], b = lg[r * 2 + 1];
  float m = fmaxf(a, b);
  float ea = __expf(a - m), eb = __expf(b - m);
  float s = ea + eb;
  out[r * 2] = ea / s; out[r * 2 + 1] = eb / s;
}

// ---------------------------------------------------------------------------
// Host orchestration
// ---------------------------------------------------------------------------
extern "C" void kernel_launch(void* const* d_in, const int* in_sizes, int n_in,
                              void* d_out, int out_size, void* d_ws, size_t ws_size,
                              hipStream_t stream) {
  (void)in_sizes; (void)n_in; (void)out_size; (void)ws_size;

  // ---- inputs (setup_inputs dict order, params flattened in insertion order)
  const float* x          = (const float*)d_in[0];
  const int*   eidx       = (const int*)d_in[1];    // [2,E] row0=src row1=dst
  const float* img_feat   = (const float*)d_in[2];
  const float* edge_emb   = (const float*)d_in[4];  // [E,1]
  const int*   tadj       = (const int*)d_in[5];
  const float* xfc_W = (const float*)d_in[8],  *xfc_b = (const float*)d_in[9];
  const float* xfc_g = (const float*)d_in[10], *xfc_be = (const float*)d_in[11];
  const float* ofc_W = (const float*)d_in[12], *ofc_b = (const float*)d_in[13];
  const float* ofc_g = (const float*)d_in[14], *ofc_be = (const float*)d_in[15];
  const float* ep_W  = (const float*)d_in[16], *ep_b = (const float*)d_in[17];
  // gs: 18..24  gt: 25..31
  const float* gat_Wl[2]  = {(const float*)d_in[18], (const float*)d_in[25]};
  const float* gat_bl[2]  = {(const float*)d_in[19], (const float*)d_in[26]};
  const float* gat_Wr[2]  = {(const float*)d_in[20], (const float*)d_in[27]};
  const float* gat_br[2]  = {(const float*)d_in[21], (const float*)d_in[28]};
  const float* gat_We[2]  = {(const float*)d_in[22], (const float*)d_in[29]};
  const float* gat_att[2] = {(const float*)d_in[23], (const float*)d_in[30]};
  const int*   gat_ei[2]  = {eidx, tadj};
  const float* p1_w = (const float*)d_in[32], *p2_w = (const float*)d_in[33];
  const float* ifc_W = (const float*)d_in[34], *ifc_b = (const float*)d_in[35];
  const float* ifc_g = (const float*)d_in[36], *ifc_be = (const float*)d_in[37];
  const float* gru0_Wih = (const float*)d_in[38], *gru0_Whh = (const float*)d_in[39];
  const float* gru0_bih = (const float*)d_in[40], *gru0_bhh = (const float*)d_in[41];
  const float* gru1_Wih = (const float*)d_in[42], *gru1_Whh = (const float*)d_in[43];
  const float* gru1_bih = (const float*)d_in[44], *gru1_bhh = (const float*)d_in[45];
  const float* mha_Wq = (const float*)d_in[46], *mha_bq = (const float*)d_in[47];
  const float* mha_Wk = (const float*)d_in[48], *mha_bk = (const float*)d_in[49];
  const float* mha_Wv = (const float*)d_in[50], *mha_bv = (const float*)d_in[51];
  const float* mha_Wo = (const float*)d_in[52], *mha_bo = (const float*)d_in[53];
  const float* cls_W1 = (const float*)d_in[54], *cls_b1 = (const float*)d_in[55];
  const float* cls_g1 = (const float*)d_in[56], *cls_be1 = (const float*)d_in[57];
  const float* cls_W2 = (const float*)d_in[58], *cls_b2 = (const float*)d_in[59];
  const float* cls_g2 = (const float*)d_in[60], *cls_be2 = (const float*)d_in[61];
  const float* cls_W3 = (const float*)d_in[62], *cls_b3 = (const float*)d_in[63];
  const float* ew_W1 = (const float*)d_in[64], *ew_b1 = (const float*)d_in[65];
  const float* ew_W2 = (const float*)d_in[66], *ew_b2 = (const float*)d_in[67];

  float* dout = (float*)d_out;   // [logits 200 | probs 200 | warn 100]

  // ---- workspace carve-up (floats) ----
  float* WS = (float*)d_ws;
  size_t off = 0;
  auto take = [&](size_t n) { float* p = WS + off; off += (n + 255) & ~(size_t)255; return p; };
  float* xc     = take((size_t)NNODE * 768);
  float* xfraw  = take((size_t)NNODE * 512);
  float* olraw  = take((size_t)NNODE * 256);
  float* eemb   = take((size_t)NEDGE * 128);
  float* eembsum= take(256);
  float* ea     = take((size_t)NEDGE * FGAT);
  float* eamean = take(512);
  float* xlb    = take((size_t)NNODE * FGAT);
  float* xrb    = take((size_t)NNODE * FGAT);
  float* gout   = take((size_t)NNODE * FGAT);
  float* nemb   = take((size_t)NNODE * 1024);
  float* elog   = take((size_t)(NEDGE + NNODE) * HEADS);
  float* ealp   = take((size_t)(NEDGE + NNODE) * HEADS);
  float* mnode  = take((size_t)NNODE * HEADS);
  float* snode  = take((size_t)NNODE * HEADS);
  float* stats  = take(4096);
  float* scbuf  = take((size_t)TGRAPH * NPG);
  float* pool1  = take((size_t)TGRAPH * K1 * 1024);
  float* pool2  = take((size_t)TGRAPH * K2 * 1024);
  float* gbuf   = take((size_t)TGRAPH * 2048);
  float* wninv  = take(64);
  float* imraw  = take((size_t)TGRAPH * 512);
  float* imn    = take((size_t)TGRAPH * 512);
  float* WihT0  = take((size_t)512 * 768);
  float* WhhT0  = take((size_t)256 * 768);
  float* WihT1  = take((size_t)256 * 768);
  float* WhhT1  = take((size_t)256 * 768);
  float* gi0    = take((size_t)TGRAPH * 768);
  float* ys0    = take((size_t)TGRAPH * 256);
  float* gi1    = take((size_t)TGRAPH * 768);
  float* ys1    = take((size_t)TGRAPH * 256);
  float* qb     = take((size_t)TGRAPH * 2048);
  float* kb     = take((size_t)TGRAPH * 2048);
  float* vb     = take((size_t)TGRAPH * 2048);
  float* aob    = take((size_t)TGRAPH * 2048);
  float* fbuf   = take((size_t)TGRAPH * 4096);
  float* h1raw  = take((size_t)TGRAPH * 256);
  float* h1     = take((size_t)TGRAPH * 256);
  float* h2raw  = take((size_t)TGRAPH * 128);
  float* h2b    = take((size_t)TGRAPH * 128);
  float* w1     = take((size_t)TGRAPH * 128);

  auto gemm = [&](const float* A, int lda, const float* B, const float* bias,
                  float* C, int ldc, int M, int N, int K) {
    dim3 grid((M + TBM - 1) / TBM, (N + TBN - 1) / TBN);
    gemm_f16_wmma<<<grid, 256, 0, stream>>>(A, lda, B, bias, C, ldc, M, N, K);
  };
  auto stats_of = [&](const float* A, int M, int N, float* sbuf) {
    (void)hipMemsetAsync(sbuf, 0, sizeof(float) * 2 * N, stream);
    dim3 grid((N + 255) / 256, 64);
    colstats<<<grid, 256, 0, stream>>>(A, M, N, sbuf);
  };
  auto napply = [&](const float* A, const float* sbuf, const float* g, const float* b,
                    float* out, int M, int N, int ostride, int ocol0, int flags) {
    size_t tot = (size_t)M * N;
    norm_apply<<<(unsigned)((tot + 255) / 256), 256, 0, stream>>>(
        A, sbuf, g, b, out, M, N, ostride, ocol0, flags);
  };

  const int EN = NEDGE + NNODE;

  // ===== node encoders =====
  gemm(x, DIN + DLBL, xfc_W, xfc_b, xfraw, 512, NNODE, 512, DIN);
  stats_of(xfraw, NNODE, 512, stats);
  napply(xfraw, stats, xfc_g, xfc_be, xc, NNODE, 512, 768, 0, 3);
  gemm(x + DIN, DIN + DLBL, ofc_W, ofc_b, olraw, 256, NNODE, 256, DLBL);
  stats_of(olraw, NNODE, 256, stats);
  napply(olraw, stats, ofc_g, ofc_be, xc, NNODE, 256, 768, 512, 3);

  // ===== edge embeddings + column sums (for self-loop mean fill) =====
  edge_embed<<<(unsigned)(((size_t)NEDGE * 128 + 255) / 256), 256, 0, stream>>>(
      edge_emb, ep_W, ep_b, eemb, NEDGE);
  stats_of(eemb, NEDGE, 128, eembsum);

  // ===== two GATv2 passes (spatial: writes nemb[:, :512]; temporal: [:,512:]) =====
  for (int gpass = 0; gpass < 2; ++gpass) {
    gemm(xc, 768, gat_Wl[gpass], gat_bl[gpass], xlb, FGAT, NNODE, FGAT, 768);
    gemm(xc, 768, gat_Wr[gpass], gat_br[gpass], xrb, FGAT, NNODE, FGAT, 768);
    gemm(eemb, 128, gat_We[gpass], nullptr, ea, FGAT, NEDGE, FGAT, 128);
    ea_mean_k<<<2, 256, 0, stream>>>(eembsum, gat_We[gpass], 1.0f / NEDGE, eamean);

    fillf<<<(unsigned)((NNODE * HEADS + 255) / 256), 256, 0, stream>>>(mnode, -3.0e38f, (size_t)NNODE * HEADS);
    (void)hipMemsetAsync(snode, 0, sizeof(float) * NNODE * HEADS, stream);
    (void)hipMemsetAsync(gout, 0, sizeof(float) * (size_t)NNODE * FGAT, stream);

    const int* srcP = gat_ei[gpass];
    const int* dstP = gat_ei[gpass] + NEDGE;
    gat_logits<<<(unsigned)((EN + 7) / 8), 256, 0, stream>>>(
        srcP, dstP, NEDGE, NNODE, xlb, xrb, ea, eamean, gat_att[gpass], elog, mnode);
    gat_expsum<<<(unsigned)(((size_t)EN * HEADS + 255) / 256), 256, 0, stream>>>(
        dstP, NEDGE, NNODE, elog, mnode, ealp, snode);
    gat_aggregate<<<EN, 256, 0, stream>>>(srcP, dstP, NEDGE, NNODE, xlb, ealp, snode, gout);
    // bias is annihilated by InstanceNorm(mean-subtract, affine=False) -> skip it
    stats_of(gout, NNODE, FGAT, stats);
    napply(gout, stats, nullptr, nullptr, nemb, NNODE, FGAT, 1024, gpass * 512, 1);
  }

  // ===== TopK pooling x2 + readout =====
  inv_norm<<<1, 256, 0, stream>>>(p1_w, 1024, wninv);
  topk_scores<<<TGRAPH, 128, 0, stream>>>(nemb, p1_w, wninv, NPG, scbuf);
  topk_select<<<TGRAPH, 256, 0, stream>>>(nemb, scbuf, NPG, K1, pool1);
  inv_norm<<<1, 256, 0, stream>>>(p2_w, 1024, wninv);
  topk_scores<<<TGRAPH, 128, 0, stream>>>(pool1, p2_w, wninv, K1, scbuf);
  topk_select<<<TGRAPH, 256, 0, stream>>>(pool1, scbuf, K1, K2, pool2);
  graph_readout<<<TGRAPH, 256, 0, stream>>>(pool2, K2, gbuf);

  // ===== image branch: FC+BN+lrelu, 2x GRU =====
  gemm(img_feat, DIN, ifc_W, ifc_b, imraw, 512, TGRAPH, 512, DIN);
  stats_of(imraw, TGRAPH, 512, stats);
  napply(imraw, stats, ifc_g, ifc_be, imn, TGRAPH, 512, 512, 0, 3);

  transpose_k<<<(unsigned)((768 * 512 + 255) / 256), 256, 0, stream>>>(gru0_Wih, WihT0, 768, 512);
  transpose_k<<<(unsigned)((768 * 256 + 255) / 256), 256, 0, stream>>>(gru0_Whh, WhhT0, 768, 256);
  transpose_k<<<(unsigned)((768 * 256 + 255) / 256), 256, 0, stream>>>(gru1_Wih, WihT1, 768, 256);
  transpose_k<<<(unsigned)((768 * 256 + 255) / 256), 256, 0, stream>>>(gru1_Whh, WhhT1, 768, 256);

  gemm(imn, 512, WihT0, gru0_bih, gi0, 768, TGRAPH, 768, 512);
  gru_seq<<<1, 256, 0, stream>>>(gi0, WhhT0, gru0_bhh, ys0, TGRAPH);
  gemm(ys0, 256, WihT1, gru1_bih, gi1, 768, TGRAPH, 768, 256);
  gru_seq<<<1, 256, 0, stream>>>(gi1, WhhT1, gru1_bhh, ys1, TGRAPH);

  // ===== cross attention =====
  gemm(gbuf, 2048, mha_Wq, mha_bq, qb, 2048, TGRAPH, 2048, 2048);
  gemm(ys1, 256, mha_Wk, mha_bk, kb, 2048, TGRAPH, 2048, 256);
  gemm(ys1, 256, mha_Wv, mha_bv, vb, 2048, TGRAPH, 2048, 256);
  cross_attn<<<dim3(TGRAPH, HEADS), 128, 0, stream>>>(qb, kb, vb, aob, TGRAPH);
  gemm(aob, 2048, mha_Wo, mha_bo, fbuf, 4096, TGRAPH, 2048, 2048);   // f[:, :2048]
  copy_strided<<<(unsigned)((TGRAPH * 2048 + 255) / 256), 256, 0, stream>>>(
      gbuf, fbuf + 2048, TGRAPH, 2048, 2048, 4096);                  // f[:, 2048:]

  // ===== classifier =====
  gemm(fbuf, 4096, cls_W1, cls_b1, h1raw, 256, TGRAPH, 256, 4096);
  stats_of(h1raw, TGRAPH, 256, stats);
  napply(h1raw, stats, cls_g1, cls_be1, h1, TGRAPH, 256, 256, 0, 3);
  gemm(h1, 256, cls_W2, cls_b2, h2raw, 128, TGRAPH, 128, 256);
  stats_of(h2raw, TGRAPH, 128, stats);
  napply(h2raw, stats, cls_g2, cls_be2, h2b, TGRAPH, 128, 128, 0, 3);
  gemm(h2b, 128, cls_W3, cls_b3, dout, 2, TGRAPH, 2, 128);           // logits
  softmax2<<<1, 128, 0, stream>>>(dout, dout + 200, TGRAPH);         // probs

  // ===== early-warning head =====
  gemm(fbuf, 4096, ew_W1, ew_b1, w1, 128, TGRAPH, 128, 4096);
  lrelu_inplace<<<(unsigned)((TGRAPH * 128 + 255) / 256), 256, 0, stream>>>(w1, (size_t)TGRAPH * 128);
  gemm(w1, 128, ew_W2, ew_b2, dout + 400, 1, TGRAPH, 1, 128);
  sigmoid_inplace<<<1, 128, 0, stream>>>(dout + 400, TGRAPH);
}
